// CollaborationGNNWithFeatures_39548058862205
// MI455X (gfx1250) — matrix operand order repaired and checked by reading, not compiled, not run on hardware
//
#include <hip/hip_runtime.h>
#include <hip/hip_bf16.h>
#include <math.h>

#define N_NODES 100000
#define E_EDGES 1600000
#define P_EDGES 1000000
#define DIN_C   128
#define H_C     128
#define DOUT_C  64
#define EIN_C   32
#define EPS_F   1e-5f

typedef __attribute__((ext_vector_type(16))) __bf16       v16bf;
typedef __attribute__((ext_vector_type(8)))  float        v8f;
typedef __attribute__((ext_vector_type(8)))  unsigned int v8u;
typedef __attribute__((ext_vector_type(4)))  unsigned int v4u;

// keep loop-variant: forbid LICM from hoisting LDS fragment loads out of the tile loop
#define TILE_CLOBBER() asm volatile("" ::: "memory")

// pack two f32 into one dword of two bf16 (RNE)
static __device__ __forceinline__ unsigned int pk2(float lo, float hi) {
  __bf16 l = (__bf16)lo, h = (__bf16)hi;
  unsigned short lb = __builtin_bit_cast(unsigned short, l);
  unsigned short hb = __builtin_bit_cast(unsigned short, h);
  return ((unsigned int)hb << 16) | (unsigned int)lb;
}

// 8 consecutive dwords from LDS, 16B-aligned -> two ds_load_b128
static __device__ __forceinline__ v16bf load_bfrag(const unsigned int* p) {
  const v4u* q = (const v4u*)p;
  v4u b0 = q[0], b1 = q[1];
  v8u bw = __builtin_shufflevector(b0, b1, 0, 1, 2, 3, 4, 5, 6, 7);
  return __builtin_bit_cast(v16bf, bw);
}

// ---------------- scatter (mean-aggregation numerator + degree) ----------------
__global__ void scatter_add_f32(const float* __restrict__ x, const int* __restrict__ src,
                                const int* __restrict__ dst, float* __restrict__ msg,
                                float* __restrict__ cnt) {
  long long t = (long long)blockIdx.x * blockDim.x + threadIdx.x;
  if (t >= (long long)E_EDGES * 32) return;
  int e = (int)(t >> 5);
  int c = (int)(t & 31) << 2;
  int s = src[e], d = dst[e];
  const float* xs = x + (size_t)s * DIN_C + c;
  float* md = msg + (size_t)d * DIN_C + c;
  atomicAdd(md + 0, xs[0]); atomicAdd(md + 1, xs[1]);
  atomicAdd(md + 2, xs[2]); atomicAdd(md + 3, xs[3]);
  if (c == 0) atomicAdd(cnt + d, 1.0f);
}

__global__ void scatter_add_bf16(const __bf16* __restrict__ h, const int* __restrict__ src,
                                 const int* __restrict__ dst, float* __restrict__ msg) {
  long long t = (long long)blockIdx.x * blockDim.x + threadIdx.x;
  if (t >= (long long)E_EDGES * 32) return;
  int e = (int)(t >> 5);
  int c = (int)(t & 31) << 2;
  int s = src[e], d = dst[e];
  const __bf16* hs = h + (size_t)s * H_C + c;
  float* md = msg + (size_t)d * H_C + c;
  atomicAdd(md + 0, (float)hs[0]); atomicAdd(md + 1, (float)hs[1]);
  atomicAdd(md + 2, (float)hs[2]); atomicAdd(md + 3, (float)hs[3]);
}

// ---------------- fused SAGEConv: out = bn_relu( (msg/cnt)@Wl^T + bias + part2@Wr^T ) ----
template<int NOUT, bool P2BF16, bool BNRELU>
__global__ void __launch_bounds__(256)
sage_gemm(const float* __restrict__ msg, const float* __restrict__ cnt,
          const void* __restrict__ part2,
          const float* __restrict__ Wl, const float* __restrict__ Wr,
          const float* __restrict__ bias,
          const float* __restrict__ bng, const float* __restrict__ bnb,
          const float* __restrict__ bnm, const float* __restrict__ bnv,
          __bf16* __restrict__ out) {
  constexpr int K = 128, KW = 64, PITCH = KW + 4, NT = NOUT / 16;
  extern __shared__ unsigned int lds[];
  for (int i = threadIdx.x; i < NOUT * KW; i += blockDim.x) {
    int n = i / KW, w = i % KW;
    lds[n * PITCH + w]                = pk2(Wl[n * K + 2 * w], Wl[n * K + 2 * w + 1]);
    lds[NOUT * PITCH + n * PITCH + w] = pk2(Wr[n * K + 2 * w], Wr[n * K + 2 * w + 1]);
  }
  __syncthreads();
  const int lane = threadIdx.x & 31, wv = threadIdx.x >> 5;
  const int half = lane >> 4, mr = lane & 15;

  // per-lane epilogue constants (tile-invariant): out = val*S + T (then ReLU)
  float S[NT], T[NT];
  #pragma unroll
  for (int nt = 0; nt < NT; ++nt) {
    const int col = nt * 16 + mr;
    if constexpr (BNRELU) {
      const float s = bng[col] * rsqrtf(bnv[col] + EPS_F);
      S[nt] = s;
      T[nt] = (bias[col] - bnm[col]) * s + bnb[col];
    } else {
      S[nt] = 1.0f;
      T[nt] = bias[col];
    }
  }

  const int tiles = N_NODES / 16;
  const int gw = (blockDim.x >> 5) * gridDim.x;
  for (int tile = blockIdx.x * (blockDim.x >> 5) + wv; tile < tiles; tile += gw) {
    TILE_CLOBBER();
    const int row = tile * 16 + mr;
    const float inv = 1.0f / fmaxf(cnt[row], 1.0f);
    const float* mrw = msg + (size_t)row * K;
    v8f acc[NT];
    #pragma unroll
    for (int nt = 0; nt < NT; ++nt) acc[nt] = v8f{};
    #pragma unroll
    for (int p = 0; p < 2; ++p) {
      #pragma unroll
      for (int kc = 0; kc < 4; ++kc) {
        v8u aw;
        #pragma unroll
        for (int j = 0; j < 8; ++j) {
          const int w = kc * 16 + half * 8 + j;
          if (p == 0) {
            aw[j] = pk2(mrw[2 * w] * inv, mrw[2 * w + 1] * inv);
          } else if (P2BF16) {
            aw[j] = ((const unsigned int*)part2)[(size_t)row * KW + w];
          } else {
            const float* xr = (const float*)part2 + (size_t)row * K;
            aw[j] = pk2(xr[2 * w], xr[2 * w + 1]);
          }
        }
        v16bf a = __builtin_bit_cast(v16bf, aw);
        #pragma unroll
        for (int nt = 0; nt < NT; ++nt) {
          v16bf bb = load_bfrag(lds + p * NOUT * PITCH +
                                (nt * 16 + mr) * PITCH + kc * 16 + half * 8);
          acc[nt] = __builtin_amdgcn_wmma_f32_16x16x32_bf16(
              false, a, false, bb, (short)0, acc[nt], false, false);
        }
      }
    }
    #pragma unroll
    for (int nt = 0; nt < NT; ++nt) {
      const int col = nt * 16 + mr;
      #pragma unroll
      for (int r = 0; r < 8; ++r) {
        const int orow = tile * 16 + half * 8 + r;
        float vv = acc[nt][r] * S[nt] + T[nt];
        if constexpr (BNRELU) vv = fmaxf(vv, 0.0f);
        out[(size_t)orow * NOUT + col] = (__bf16)vv;
      }
    }
  }
}

// ---------------- generic row-tiled MLP GEMM (A row-major, W [NOUT][K] f32) ------------
template<int K, int NT, bool A32, bool BNRELU>
__global__ void __launch_bounds__(256)
mlp_gemm(const void* __restrict__ A, int strideA,
         const float* __restrict__ W, const float* __restrict__ bias,
         const float* __restrict__ bng, const float* __restrict__ bnb,
         const float* __restrict__ bnm, const float* __restrict__ bnv,
         __bf16* __restrict__ out, int strideO, int colOff, int rows) {
  constexpr int KW = K / 2, PITCH = KW + 4, KCH = K / 32, NOUT = NT * 16;
  extern __shared__ unsigned int lds[];
  for (int i = threadIdx.x; i < NOUT * KW; i += blockDim.x) {
    int n = i / KW, w = i % KW;
    lds[n * PITCH + w] = pk2(W[n * K + 2 * w], W[n * K + 2 * w + 1]);
  }
  __syncthreads();
  const int lane = threadIdx.x & 31, wv = threadIdx.x >> 5;
  const int half = lane >> 4, mr = lane & 15;

  float S[NT], T[NT];
  #pragma unroll
  for (int nt = 0; nt < NT; ++nt) {
    const int col = nt * 16 + mr;
    if constexpr (BNRELU) {
      const float s = bng[col] * rsqrtf(bnv[col] + EPS_F);
      S[nt] = s;
      T[nt] = (bias[col] - bnm[col]) * s + bnb[col];
    } else {
      S[nt] = 1.0f;
      T[nt] = bias[col];
    }
  }

  const int tiles = (rows + 15) >> 4;
  const int gw = (blockDim.x >> 5) * gridDim.x;
  for (int tile = blockIdx.x * (blockDim.x >> 5) + wv; tile < tiles; tile += gw) {
    if constexpr (KCH * NT > 8) TILE_CLOBBER();   // big kernels: keep B in LDS per tile
    int row = tile * 16 + mr;
    int rc = row < rows ? row : rows - 1;
    v8f acc[NT];
    #pragma unroll
    for (int nt = 0; nt < NT; ++nt) acc[nt] = v8f{};
    #pragma unroll
    for (int kc = 0; kc < KCH; ++kc) {
      v8u aw;
      #pragma unroll
      for (int j = 0; j < 8; ++j) {
        const int w = kc * 16 + half * 8 + j;
        if (A32) {
          const float* ar = (const float*)A + (size_t)rc * strideA;
          aw[j] = pk2(ar[2 * w], ar[2 * w + 1]);
        } else {
          aw[j] = ((const unsigned int*)A)[(size_t)rc * (strideA >> 1) + w];
        }
      }
      v16bf a = __builtin_bit_cast(v16bf, aw);
      #pragma unroll
      for (int nt = 0; nt < NT; ++nt) {
        v16bf bb = load_bfrag(lds + (nt * 16 + mr) * PITCH + kc * 16 + half * 8);
        acc[nt] = __builtin_amdgcn_wmma_f32_16x16x32_bf16(
            false, a, false, bb, (short)0, acc[nt], false, false);
      }
    }
    #pragma unroll
    for (int nt = 0; nt < NT; ++nt) {
      const int col = nt * 16 + mr;
      #pragma unroll
      for (int r = 0; r < 8; ++r) {
        const int orow = tile * 16 + half * 8 + r;
        float vv = acc[nt][r] * S[nt] + T[nt];
        if constexpr (BNRELU) vv = fmaxf(vv, 0.0f);
        if (orow < rows) out[(size_t)orow * strideO + colOff + col] = (__bf16)vv;
      }
    }
  }
}

// ------- predictor GEMM 1 with fused gather: A = [z[src] | z[dst] | e_out], K=160 -------
__global__ void __launch_bounds__(256)
pred1_gemm(const __bf16* __restrict__ z, const int* __restrict__ pe,
           const __bf16* __restrict__ eo,
           const float* __restrict__ W, const float* __restrict__ bias,
           const float* __restrict__ bng, const float* __restrict__ bnb,
           const float* __restrict__ bnm, const float* __restrict__ bnv,
           __bf16* __restrict__ out) {
  constexpr int K = 160, KW = 80, PITCH = KW + 4, KCH = 5, NT = 8, NOUT = 128;
  extern __shared__ unsigned int lds[];
  for (int i = threadIdx.x; i < NOUT * KW; i += blockDim.x) {
    int n = i / KW, w = i % KW;
    lds[n * PITCH + w] = pk2(W[n * K + 2 * w], W[n * K + 2 * w + 1]);
  }
  __syncthreads();
  const int lane = threadIdx.x & 31, wv = threadIdx.x >> 5;
  const int half = lane >> 4, mr = lane & 15;

  float S[NT], T[NT];
  #pragma unroll
  for (int nt = 0; nt < NT; ++nt) {
    const int col = nt * 16 + mr;
    const float s = bng[col] * rsqrtf(bnv[col] + EPS_F);
    S[nt] = s;
    T[nt] = (bias[col] - bnm[col]) * s + bnb[col];
  }

  const int tiles = P_EDGES / 16;
  const int gw = (blockDim.x >> 5) * gridDim.x;
  for (int tile = blockIdx.x * (blockDim.x >> 5) + wv; tile < tiles; tile += gw) {
    TILE_CLOBBER();
    const int row = tile * 16 + mr;
    const int sp = pe[row], dp = pe[P_EDGES + row];
    const unsigned int* zs = (const unsigned int*)(z + (size_t)sp * 64);
    const unsigned int* zd = (const unsigned int*)(z + (size_t)dp * 64);
    const unsigned int* er = (const unsigned int*)(eo + (size_t)row * 32);
    v8f acc[NT];
    #pragma unroll
    for (int nt = 0; nt < NT; ++nt) acc[nt] = v8f{};
    #pragma unroll
    for (int kc = 0; kc < KCH; ++kc) {
      v8u aw;
      #pragma unroll
      for (int j = 0; j < 8; ++j) {
        const int w = kc * 16 + half * 8 + j;
        aw[j] = (w < 32) ? zs[w] : (w < 64) ? zd[w - 32] : er[w - 64];
      }
      v16bf a = __builtin_bit_cast(v16bf, aw);
      #pragma unroll
      for (int nt = 0; nt < NT; ++nt) {
        v16bf bb = load_bfrag(lds + (nt * 16 + mr) * PITCH + kc * 16 + half * 8);
        acc[nt] = __builtin_amdgcn_wmma_f32_16x16x32_bf16(
            false, a, false, bb, (short)0, acc[nt], false, false);
      }
    }
    #pragma unroll
    for (int nt = 0; nt < NT; ++nt) {
      const int col = nt * 16 + mr;
      #pragma unroll
      for (int r = 0; r < 8; ++r) {
        const int orow = tile * 16 + half * 8 + r;
        float vv = fmaxf(acc[nt][r] * S[nt] + T[nt], 0.0f);
        out[(size_t)orow * NOUT + col] = (__bf16)vv;
      }
    }
  }
}

// ---------------- final 64->1 projection ----------------
__global__ void pred3(const __bf16* __restrict__ u2, const float* __restrict__ w3,
                      const float* __restrict__ b3, float* __restrict__ out) {
  int p = blockIdx.x * blockDim.x + threadIdx.x;
  if (p >= P_EDGES) return;
  const __bf16* ur = u2 + (size_t)p * 64;
  float s = 0.0f;
  #pragma unroll
  for (int i = 0; i < 64; ++i) s += (float)ur[i] * w3[i];
  out[p] = s + b3[0];
}

extern "C" void kernel_launch(void* const* d_in, const int* in_sizes, int n_in,
                              void* d_out, int out_size, void* d_ws, size_t ws_size,
                              hipStream_t stream) {
  (void)in_sizes; (void)n_in; (void)out_size; (void)ws_size;
  const float* x    = (const float*)d_in[0];
  const int*   ei   = (const int*)d_in[1];
  const int*   pei  = (const int*)d_in[2];
  const float* ef   = (const float*)d_in[3];
  const float* s1Wl = (const float*)d_in[4];
  const float* s1bl = (const float*)d_in[5];
  const float* s1Wr = (const float*)d_in[6];
  const float* bn1g = (const float*)d_in[7];
  const float* bn1b = (const float*)d_in[8];
  const float* bn1m = (const float*)d_in[9];
  const float* bn1v = (const float*)d_in[10];
  const float* s2Wl = (const float*)d_in[11];
  const float* s2bl = (const float*)d_in[12];
  const float* s2Wr = (const float*)d_in[13];
  const float* eW1  = (const float*)d_in[14];
  const float* eb1  = (const float*)d_in[15];
  const float* ebg  = (const float*)d_in[16];
  const float* ebb  = (const float*)d_in[17];
  const float* ebm  = (const float*)d_in[18];
  const float* ebv  = (const float*)d_in[19];
  const float* eW2  = (const float*)d_in[20];
  const float* eb2  = (const float*)d_in[21];
  const float* pW1  = (const float*)d_in[22];
  const float* pb1  = (const float*)d_in[23];
  const float* p1g  = (const float*)d_in[24];
  const float* p1b  = (const float*)d_in[25];
  const float* p1m  = (const float*)d_in[26];
  const float* p1v  = (const float*)d_in[27];
  const float* pW2  = (const float*)d_in[28];
  const float* pb2  = (const float*)d_in[29];
  const float* p2g  = (const float*)d_in[30];
  const float* p2b  = (const float*)d_in[31];
  const float* p2m  = (const float*)d_in[32];
  const float* p2v  = (const float*)d_in[33];
  const float* pW3  = (const float*)d_in[34];
  const float* pb3  = (const float*)d_in[35];
  float* outp = (float*)d_out;

  char* ws = (char*)d_ws;
  size_t off = 0;
  auto wsa = [&](size_t bytes) {
    void* p = ws + off;
    off = (off + bytes + 255) & ~(size_t)255;
    return p;
  };
  float*  msg = (float*)wsa((size_t)N_NODES * 128 * 4);
  float*  cnt = (float*)wsa((size_t)N_NODES * 4);
  __bf16* hbf = (__bf16*)wsa((size_t)N_NODES * 128 * 2);
  __bf16* zbf = (__bf16*)wsa((size_t)N_NODES * 64 * 2);
  __bf16* eo  = (__bf16*)wsa((size_t)P_EDGES * 32 * 2);
  __bf16* u1  = (__bf16*)wsa((size_t)P_EDGES * 128 * 2);   // also holds edge-hidden (dead before u1 written)
  __bf16* u2  = (__bf16*)wsa((size_t)P_EDGES * 64 * 2);
  __bf16* eh  = u1;

  const int* esrc = ei;
  const int* edst = ei + E_EDGES;
  const int sb = (int)(((long long)E_EDGES * 32 + 255) / 256);

  // SAGE layer 1
  hipMemsetAsync(msg, 0, (size_t)N_NODES * 128 * 4 + (size_t)N_NODES * 4, stream); // msg + cnt
  scatter_add_f32<<<sb, 256, 0, stream>>>(x, esrc, edst, msg, cnt);
  sage_gemm<128, false, true><<<784, 256, 2 * 128 * 68 * 4, stream>>>(
      msg, cnt, x, s1Wl, s1Wr, s1bl, bn1g, bn1b, bn1m, bn1v, hbf);

  // SAGE layer 2 (cnt reused)
  hipMemsetAsync(msg, 0, (size_t)N_NODES * 128 * 4, stream);
  scatter_add_bf16<<<sb, 256, 0, stream>>>(hbf, esrc, edst, msg);
  sage_gemm<64, true, false><<<784, 256, 2 * 64 * 68 * 4, stream>>>(
      msg, cnt, hbf, s2Wl, s2Wr, s2bl, nullptr, nullptr, nullptr, nullptr, zbf);

  // edge feature MLP
  mlp_gemm<32, 4, true, true><<<1024, 256, 64 * 20 * 4, stream>>>(
      ef, 32, eW1, eb1, ebg, ebb, ebm, ebv, eh, 64, 0, P_EDGES);
  mlp_gemm<64, 2, false, false><<<1024, 256, 32 * 36 * 4, stream>>>(
      eh, 64, eW2, eb2, nullptr, nullptr, nullptr, nullptr, eo, 32, 0, P_EDGES);

  // predictor
  pred1_gemm<<<1024, 256, 128 * 84 * 4, stream>>>(
      zbf, pei, eo, pW1, pb1, p1g, p1b, p1m, p1v, u1);
  mlp_gemm<128, 4, false, true><<<1024, 256, 64 * 68 * 4, stream>>>(
      u1, 128, pW2, pb2, p2g, p2b, p2m, p2v, u2, 64, 0, P_EDGES);
  pred3<<<(P_EDGES + 255) / 256, 256, 0, stream>>>(u2, pW3, pb3, outp);
}
